// QLSTM_58660663329238
// MI455X (gfx1250) — compile-verified, matrix operand order checked by reading, not run on hardware
//
#include <hip/hip_runtime.h>
#include <hip/hip_bf16.h>

typedef __attribute__((ext_vector_type(16))) __bf16 v16bf;
typedef __attribute__((ext_vector_type(8)))  __bf16 v8bf;
typedef __attribute__((ext_vector_type(2)))  __bf16 v2bf;
typedef __attribute__((ext_vector_type(8)))  float  v8f;
typedef __attribute__((ext_vector_type(4)))  float  v4f;

#define BQ 8
#define TQ 4096
#define DQ 1024
#define SQ 1024
#define MROWS (BQ * TQ)          // 32768
#define KTILES (DQ / 32)         // 32

__device__ __forceinline__ float fsigmoid(float x) {
    return 1.0f / (1.0f + __expf(-x));
}
__device__ __forceinline__ float ftanh(float x) {
    return 2.0f / (1.0f + __expf(-2.0f * x)) - 1.0f;
}

// ---------------------------------------------------------------------------
// Kernel A: fused quad-GEMM  Z{h,f,i,c} = X @ W{h,f,i,c}  (+bias, activation)
//   y  = Zh                         -> d_out (f32)
//   f  = sigmoid(Zf + bf)           -> ws    (bf16)
//   u  = sigmoid(Zi+bi)*tanh(Zc+bc) -> ws    (bf16)
// WG tile 64(M) x 128(N), 8 waves, wave tile 32x32 per GEMM.
// Register-buffered global fetch + LDS double buffer: one barrier per K-step,
// next tile's global_load_b128s fly underneath the 16-WMMA burst.
// ---------------------------------------------------------------------------
__global__ __launch_bounds__(256)
void qlstm_gemm_gates(const float* __restrict__ X,
                      const float* __restrict__ Wh,
                      const float* __restrict__ Wf,
                      const float* __restrict__ Wi,
                      const float* __restrict__ Wc,
                      const float* __restrict__ bF,
                      const float* __restrict__ bI,
                      const float* __restrict__ bC,
                      float* __restrict__ y,
                      __bf16* __restrict__ fOut,
                      __bf16* __restrict__ uOut)
{
    // per buffer: X tile 64x32 (2048) + 4 W tiles 128x32 transposed (16384)
    __shared__ __bf16 lds[2][18432];

    const int tid  = threadIdx.x;
    const int lane = tid & 31;
    const int wave = tid >> 5;
    const int wm   = wave & 1;                 // 2 waves along M
    const int wn   = wave >> 1;                // 4 waves along N

    const int m0 = blockIdx.y * 64;            // grid.y = 512
    const int n0 = blockIdx.x * 128;           // grid.x = 8

    const int row  = lane & 15;
    const int half = lane >> 4;

    // --- staging assignments (constant per thread) ---
    const int xm  = tid >> 2;                  // X: row 0..63
    const int xk0 = (tid & 3) * 8;             // X: k 0,8,16,24
    int kzA[2], n4A[2];                        // W: 2 tasks/thread/gemm
#pragma unroll
    for (int i = 0; i < 2; ++i) {
        const int q = i * 256 + tid;           // 512 (k-pair, n4) tasks
        kzA[i] = q >> 5;                       // k-pair 0..15
        n4A[i] = (q & 31) * 4;                 // n 0..124 step 4
    }
    const float* Wg[4] = {Wh, Wf, Wi, Wc};

    // staging registers
    v4f xr[2];
    v4f wr[4][2][2];                           // [gemm][task][k-row 0/1]

    auto loadRegs = [&](int kb) {
        const float* src = X + (size_t)(m0 + xm) * DQ + kb + xk0;
        xr[0] = *reinterpret_cast<const v4f*>(src);
        xr[1] = *reinterpret_cast<const v4f*>(src + 4);
#pragma unroll
        for (int g = 0; g < 4; ++g) {
#pragma unroll
            for (int i = 0; i < 2; ++i) {
                const float* w = Wg[g] + (size_t)(kb + 2 * kzA[i]) * SQ + n0 + n4A[i];
                wr[g][i][0] = *reinterpret_cast<const v4f*>(w);
                wr[g][i][1] = *reinterpret_cast<const v4f*>(w + SQ);
            }
        }
    };

    auto storeRegs = [&](__bf16* buf) {
        // X tile: packed b128 store
        v8bf v;
#pragma unroll
        for (int j = 0; j < 4; ++j) { v[j] = (__bf16)xr[0][j]; v[4 + j] = (__bf16)xr[1][j]; }
        *reinterpret_cast<v8bf*>(&buf[xm * 32 + xk0]) = v;
        // W tiles: transposed [n][k], (k,k+1) packed -> ds_store_b32
#pragma unroll
        for (int g = 0; g < 4; ++g) {
            __bf16* dst = buf + 2048 + g * 4096;
#pragma unroll
            for (int i = 0; i < 2; ++i) {
                const int kk0 = 2 * kzA[i];
#pragma unroll
                for (int j = 0; j < 4; ++j) {
                    v2bf p;
                    p[0] = (__bf16)wr[g][i][0][j];
                    p[1] = (__bf16)wr[g][i][1][j];
                    *reinterpret_cast<v2bf*>(&dst[(n4A[i] + j) * 32 + kk0]) = p;
                }
            }
        }
    };

    // fragment load per documented 16-bit A layout:
    //  lanes 0-15: VGPR0-3 = k0..7,  VGPR4-7 = k16..23
    //  lanes16-31: VGPR0-3 = k8..15, VGPR4-7 = k24..31
    auto frag = [&](const __bf16* base) -> v16bf {
        const v8bf* p = reinterpret_cast<const v8bf*>(base + row * 32 + half * 8);
        v8bf lo = p[0];
        v8bf hi = p[2];
        return __builtin_shufflevector(lo, hi,
            0, 1, 2, 3, 4, 5, 6, 7, 8, 9, 10, 11, 12, 13, 14, 15);
    };

    v8f acc[4][2][2] = {};                     // [gemm][mi][ni]

    loadRegs(0);
    for (int kt = 0; kt < KTILES; ++kt) {
        __bf16* buf = lds[kt & 1];
        storeRegs(buf);
        __syncthreads();
        if (kt + 1 < KTILES) loadRegs((kt + 1) * 32);   // overlap with WMMAs

        v16bf afr[2];
        afr[0] = frag(&buf[(wm * 32 + 0)  * 32]);
        afr[1] = frag(&buf[(wm * 32 + 16) * 32]);
#pragma unroll
        for (int g = 0; g < 4; ++g) {
#pragma unroll
            for (int ni = 0; ni < 2; ++ni) {
                v16bf bfr = frag(&buf[2048 + g * 4096 + (wn * 32 + ni * 16) * 32]);
#pragma unroll
                for (int mi = 0; mi < 2; ++mi) {
                    acc[g][mi][ni] = __builtin_amdgcn_wmma_f32_16x16x32_bf16(
                        false, afr[mi], false, bfr,
                        (short)0, acc[g][mi][ni], false, false);
                }
            }
        }
    }

    // ---- epilogue: C layout VGPR r -> M = r + (lane>=16 ? 8 : 0), N = lane&15
#pragma unroll
    for (int mi = 0; mi < 2; ++mi) {
#pragma unroll
        for (int ni = 0; ni < 2; ++ni) {
            const int mB = m0 + wm * 32 + mi * 16 + half * 8;
            const int nn = n0 + wn * 32 + ni * 16 + row;
            const float bf_v = bF[nn];
            const float bi_v = bI[nn];
            const float bc_v = bC[nn];
#pragma unroll
            for (int r = 0; r < 8; ++r) {
                const size_t off = (size_t)(mB + r) * SQ + nn;
                y[off] = acc[0][mi][ni][r];
                float fv = fsigmoid(acc[1][mi][ni][r] + bf_v);
                float uv = fsigmoid(acc[2][mi][ni][r] + bi_v) *
                           ftanh  (acc[3][mi][ni][r] + bc_v);
                fOut[off] = (__bf16)fv;
                uOut[off] = (__bf16)uv;
            }
        }
    }
}

// ---------------------------------------------------------------------------
// Kernel B: per-chunk scan. Each thread folds 512 timesteps of TWO adjacent
// (b,s) channels into affine pairs (F = prod f, U = scan-from-zero).
// dword loads, two independent FMA chains. 8 chunks x 8192 channels.
// ---------------------------------------------------------------------------
__global__ __launch_bounds__(256)
void qlstm_scan_chunks(const __bf16* __restrict__ fIn,
                       const __bf16* __restrict__ uIn,
                       float* __restrict__ chunkF,
                       float* __restrict__ chunkU)
{
    const int idx   = blockIdx.x * blockDim.x + threadIdx.x;   // 32768
    const int pair  = idx & 4095;          // b*512 + s/2  (coalesced)
    const int chunk = idx >> 12;           // 0..7
    const int b     = pair >> 9;
    const int s     = (pair & 511) * 2;
    const int t0    = chunk * 512;

    float F0 = 1.0f, U0 = 0.0f, F1 = 1.0f, U1 = 0.0f;
    size_t base = ((size_t)b * TQ + t0) * SQ + s;
    for (int t = 0; t < 512; ++t) {
        v2bf fv = *reinterpret_cast<const v2bf*>(fIn + base);
        v2bf uv = *reinterpret_cast<const v2bf*>(uIn + base);
        float f0 = (float)fv[0], f1 = (float)fv[1];
        U0 = f0 * U0 + (float)uv[0];
        U1 = f1 * U1 + (float)uv[1];
        F0 *= f0;
        F1 *= f1;
        base += SQ;
    }
    const int ch = b * 1024 + s;
    chunkF[chunk * 8192 + ch]     = F0;
    chunkF[chunk * 8192 + ch + 1] = F1;
    chunkU[chunk * 8192 + ch]     = U0;
    chunkU[chunk * 8192 + ch + 1] = U1;
}

// ---------------------------------------------------------------------------
// Kernel C: compose the 8 chunk affines with state_c -> new_c (last state).
// ---------------------------------------------------------------------------
__global__ __launch_bounds__(256)
void qlstm_scan_combine(const float* __restrict__ chunkF,
                        const float* __restrict__ chunkU,
                        const float* __restrict__ state_c,
                        float* __restrict__ newC)
{
    const int ch = blockIdx.x * blockDim.x + threadIdx.x;      // 8192
    float c = state_c[ch];
#pragma unroll
    for (int k = 0; k < 8; ++k) {
        c = chunkF[k * 8192 + ch] * c + chunkU[k * 8192 + ch];
    }
    newC[ch] = c;
}

extern "C" void kernel_launch(void* const* d_in, const int* in_sizes, int n_in,
                              void* d_out, int out_size, void* d_ws, size_t ws_size,
                              hipStream_t stream)
{
    // setup_inputs order: inputs, state_c, Wc, bc, Wf, bf, Wi, bi, Wo, bo, Wh
    const float* X   = (const float*)d_in[0];
    const float* sc  = (const float*)d_in[1];
    const float* Wc  = (const float*)d_in[2];
    const float* bc  = (const float*)d_in[3];
    const float* Wf  = (const float*)d_in[4];
    const float* bf  = (const float*)d_in[5];
    const float* Wi  = (const float*)d_in[6];
    const float* bi  = (const float*)d_in[7];
    // d_in[8] = Wo, d_in[9] = bo : dead code in the reference, skipped.
    const float* Wh  = (const float*)d_in[10];

    float* y    = (float*)d_out;                              // B*T*D
    float* newC = (float*)d_out + (size_t)MROWS * DQ;         // + B*S

    char* ws = (char*)d_ws;
    __bf16* fBuf = (__bf16*)ws;                               // 67 MB
    __bf16* uBuf = (__bf16*)(ws + (size_t)MROWS * SQ * 2);    // 67 MB
    float* chF = (float*)(ws + (size_t)MROWS * SQ * 4);       // 256 KB
    float* chU = chF + 65536;                                 // 256 KB

    dim3 gridA(SQ / 128, MROWS / 64, 1);                      // 8 x 512
    qlstm_gemm_gates<<<gridA, 256, 0, stream>>>(
        X, Wh, Wf, Wi, Wc, bf, bi, bc, y, fBuf, uBuf);

    qlstm_scan_chunks<<<32768 / 256, 256, 0, stream>>>(fBuf, uBuf, chF, chU);

    qlstm_scan_combine<<<8192 / 256, 256, 0, stream>>>(chF, chU, sc, newC);
}
